// SelfAttentionModule_65481071400159
// MI455X (gfx1250) — compile-verified, matrix operand order checked by reading, not run on hardware
//
#include <hip/hip_runtime.h>

// ---------------------------------------------------------------------------
// Problem constants (from the reference)
// ---------------------------------------------------------------------------
#define E_DIM 1024
#define H_DIM 16
#define D_DIM 64
#define B_DIM 2
#define S_DIM 2048
#define M_DIM (B_DIM * S_DIM)   // 4096 rows total
#define N_QKV (3 * E_DIM)       // 3072

typedef __attribute__((ext_vector_type(16))) __bf16 v16bf;
typedef __attribute__((ext_vector_type(8)))  __bf16 v8bf;
typedef __attribute__((ext_vector_type(8)))  float  v8f;
typedef __attribute__((ext_vector_type(4)))  int    v4i;

// ---------------------------------------------------------------------------
// CDNA5 async global->LDS copy (ASYNCcnt path), guarded so a toolchain
// without the builtin falls back to the load/ds_store path.
// Builtin signature (from hipcc diagnostic): param0 = <4 x i32>* (global),
// param1 = <4 x i32>* (LDS), then imm offset, imm cpol.
// ---------------------------------------------------------------------------
#if defined(__has_builtin)
#if __has_builtin(__builtin_amdgcn_global_load_async_to_lds_b128) && \
    __has_builtin(__builtin_amdgcn_s_wait_asynccnt)
#define USE_ASYNC 1
#endif
#endif
#ifndef USE_ASYNC
#define USE_ASYNC 0
#endif

#if USE_ASYNC
typedef __attribute__((address_space(1))) v4i glb_v4i;
typedef __attribute__((address_space(3))) v4i lds_v4i;
__device__ __forceinline__ void async_copy16(const __bf16* g, __bf16* l) {
  __builtin_amdgcn_global_load_async_to_lds_b128(
      (glb_v4i*)(g), (lds_v4i*)(l), 0, 0);
}
#define ASYNC_WAIT() __builtin_amdgcn_s_wait_asynccnt(0)
#else
#define ASYNC_WAIT() ((void)0)
#endif

// ---------------------------------------------------------------------------
// fp32 -> bf16 (round-to-nearest-even-ish)
// ---------------------------------------------------------------------------
__device__ __forceinline__ __bf16 f2bf(float f) {
  unsigned u = __builtin_bit_cast(unsigned, f);
  u += 0x7FFFu + ((u >> 16) & 1u);
  unsigned short h = (unsigned short)(u >> 16);
  return __builtin_bit_cast(__bf16, h);
}

// ---------------------------------------------------------------------------
// Load a 16x32 bf16 WMMA A/B fragment from an LDS tile (row-major, STRIDE
// elements per row).  lane&15 = row; K halves split across half-waves per the
// CDNA5 16-bit operand layout.
// ---------------------------------------------------------------------------
template<int STRIDE>
__device__ __forceinline__ v16bf load_frag(const __bf16* base, int lane) {
  const int r  = lane & 15;
  const int kb = (lane >> 4) * 8;
  const __bf16* p = base + r * STRIDE + kb;
  v8bf lo = *(const v8bf*)(p);
  v8bf hi = *(const v8bf*)(p + 16);
  return __builtin_shufflevector(lo, hi,
      0, 1, 2, 3, 4, 5, 6, 7, 8, 9, 10, 11, 12, 13, 14, 15);
}

// ---------------------------------------------------------------------------
// Conversion kernels
// ---------------------------------------------------------------------------
__global__ void convert_bf16_kernel(const float* __restrict__ src,
                                    __bf16* __restrict__ dst, int n) {
  int i = (blockIdx.x * blockDim.x + threadIdx.x) * 8;
  if (i + 7 < n) {
    float4 f0 = *(const float4*)(src + i);
    float4 f1 = *(const float4*)(src + i + 4);
    v8bf o;
    o[0] = f2bf(f0.x); o[1] = f2bf(f0.y); o[2] = f2bf(f0.z); o[3] = f2bf(f0.w);
    o[4] = f2bf(f1.x); o[5] = f2bf(f1.y); o[6] = f2bf(f1.z); o[7] = f2bf(f1.w);
    *(v8bf*)(dst + i) = o;
  }
}

// dst[c*rows + r] = src[r*cols + c]  (produces B^T layout for q@R / k@Ent)
__global__ void transpose_bf16_kernel(const float* __restrict__ src,
                                      __bf16* __restrict__ dst,
                                      int rows, int cols) {
  int c = blockIdx.x * blockDim.x + threadIdx.x;
  int r = blockIdx.y;
  if (c < cols) dst[(size_t)c * rows + r] = f2bf(src[(size_t)r * cols + c]);
}

// ---------------------------------------------------------------------------
// Generic bf16 GEMM:  C[M,N] = A[M,K] * Bt[N,K]^T  (+bias)  (+residual, f32)
// 256 threads (8 waves), 128x128 C tile, K-step 32, double-buffered LDS.
// Wave (wm 0..1, wn 0..3) owns a 64x32 sub-tile = 4x2 WMMA accumulators.
// ---------------------------------------------------------------------------
#define LSTR 48   // LDS row stride in bf16 elements (96B rows, 16B aligned)
#define BKK  32

template<bool F32OUT>
__global__ __launch_bounds__(256)
void gemm_bf16(const __bf16* __restrict__ A, int lda,
               const __bf16* __restrict__ Bt, int ldb,
               void* __restrict__ Cout, int ldc, int K,
               const float* __restrict__ bias,
               const float* __restrict__ residual) {
  __shared__ __align__(16) __bf16 As[2][128 * LSTR];
  __shared__ __align__(16) __bf16 Bs[2][128 * LSTR];

  const int tid  = threadIdx.x;
  const int lane = tid & 31;
  const int wid  = tid >> 5;
  const int wm   = wid >> 2;   // 0..1
  const int wn   = wid & 3;    // 0..3
  const int m0   = blockIdx.y * 128;
  const int n0   = blockIdx.x * 128;

  v8f acc[4][2] = {};

  const int lrow = tid >> 1;          // 0..127 : tile row this thread stages
  const int lseg = (tid & 1) * 16;    // 0 / 16 : 16-element column segment
  const int lofs = lrow * LSTR + lseg;
  const __bf16* gA = A  + (size_t)(m0 + lrow) * lda + lseg;
  const __bf16* gB = Bt + (size_t)(n0 + lrow) * ldb + lseg;

  int buf = 0;
#if !USE_ASYNC
  v8bf ra0, ra1, rb0, rb1;
#endif

  // -------- prologue: stage K-step 0 into buffer 0 --------
#if USE_ASYNC
  async_copy16(gA,     &As[0][lofs]);
  async_copy16(gA + 8, &As[0][lofs + 8]);
  async_copy16(gB,     &Bs[0][lofs]);
  async_copy16(gB + 8, &Bs[0][lofs + 8]);
#else
  ra0 = *(const v8bf*)(gA);
  ra1 = *(const v8bf*)(gA + 8);
  rb0 = *(const v8bf*)(gB);
  rb1 = *(const v8bf*)(gB + 8);
  *(v8bf*)(&As[0][lofs])     = ra0;
  *(v8bf*)(&As[0][lofs + 8]) = ra1;
  *(v8bf*)(&Bs[0][lofs])     = rb0;
  *(v8bf*)(&Bs[0][lofs + 8]) = rb1;
#endif
  ASYNC_WAIT();
  __syncthreads();

  for (int k0 = 0; k0 < K; k0 += BKK) {
    const bool have_next = (k0 + BKK < K);
    if (have_next) {
#if USE_ASYNC
      async_copy16(gA + k0 + BKK,     &As[buf ^ 1][lofs]);
      async_copy16(gA + k0 + BKK + 8, &As[buf ^ 1][lofs + 8]);
      async_copy16(gB + k0 + BKK,     &Bs[buf ^ 1][lofs]);
      async_copy16(gB + k0 + BKK + 8, &Bs[buf ^ 1][lofs + 8]);
#else
      ra0 = *(const v8bf*)(gA + k0 + BKK);
      ra1 = *(const v8bf*)(gA + k0 + BKK + 8);
      rb0 = *(const v8bf*)(gB + k0 + BKK);
      rb1 = *(const v8bf*)(gB + k0 + BKK + 8);
#endif
      if (k0 + 2 * BKK < K) {
        __builtin_prefetch(gA + k0 + 2 * BKK, 0, 3);
        __builtin_prefetch(gB + k0 + 2 * BKK, 0, 3);
      }
    }

    // -------- compute on current buffer --------
    {
      v16bf a[4], b[2];
#pragma unroll
      for (int i = 0; i < 4; ++i)
        a[i] = load_frag<LSTR>(&As[buf][(wm * 64 + i * 16) * LSTR], lane);
#pragma unroll
      for (int j = 0; j < 2; ++j)
        b[j] = load_frag<LSTR>(&Bs[buf][(wn * 32 + j * 16) * LSTR], lane);
#pragma unroll
      for (int i = 0; i < 4; ++i)
#pragma unroll
        for (int j = 0; j < 2; ++j)
          acc[i][j] = __builtin_amdgcn_wmma_f32_16x16x32_bf16(
              false, a[i], false, b[j], (short)0, acc[i][j], false, false);
    }

#if !USE_ASYNC
    if (have_next) {   // buf^1 was last read 1 iteration ago; safe to refill
      *(v8bf*)(&As[buf ^ 1][lofs])     = ra0;
      *(v8bf*)(&As[buf ^ 1][lofs + 8]) = ra1;
      *(v8bf*)(&Bs[buf ^ 1][lofs])     = rb0;
      *(v8bf*)(&Bs[buf ^ 1][lofs + 8]) = rb1;
    }
#endif
    ASYNC_WAIT();
    __syncthreads();
    buf ^= 1;
  }

  // Epilogue: C-layout element e -> row e + 8*(lane>=16), col = lane&15
  const int cl    = lane & 15;
  const int half8 = (lane >> 4) * 8;
#pragma unroll
  for (int i = 0; i < 4; ++i) {
#pragma unroll
    for (int j = 0; j < 2; ++j) {
      const int col = n0 + wn * 32 + j * 16 + cl;
      const float bv = bias ? bias[col] : 0.0f;
#pragma unroll
      for (int e = 0; e < 8; ++e) {
        const int row = m0 + wm * 64 + i * 16 + half8 + e;
        float v = acc[i][j][e] + bv;
        if (F32OUT) {
          ((float*)Cout)[(size_t)row * ldc + col] =
              v + residual[(size_t)row * ldc + col];
        } else {
          ((__bf16*)Cout)[(size_t)row * ldc + col] = f2bf(v);
        }
      }
    }
  }
}

// ---------------------------------------------------------------------------
// Flash-attention kernel.  One block = one (b, h, 128 q-rows) tile.
// 8 waves; wave w owns q rows [w*16, w*16+16).  Streams 64-key blocks with
// double-buffered K / V^T tiles (K tile goes through the async-LDS path).
// ---------------------------------------------------------------------------
#define QSTR 72   // LDS row stride (144B rows, 16B aligned)

__global__ __launch_bounds__(256)
void attn_kernel(const __bf16* __restrict__ Q,   // (M, E) row-major
                 const __bf16* __restrict__ Kk,  // (M, E) row-major
                 const __bf16* __restrict__ V,   // qkv buffer + 2E, stride 3E
                 __bf16* __restrict__ O) {       // (M, E) row-major
  __shared__ __align__(16) __bf16 Qs [128 * QSTR];
  __shared__ __align__(16) __bf16 Ks [2][64 * QSTR];
  __shared__ __align__(16) __bf16 Vts[2][64 * QSTR];  // transposed: [d][key]
  __shared__ __align__(16) __bf16 Ps [8][16 * QSTR];  // wave-private P tiles

  const int tid  = threadIdx.x;
  const int lane = tid & 31;
  const int wid  = tid >> 5;
  const int bh   = blockIdx.y;          // b*H + h
  const int bN   = bh >> 4;
  const int h    = bh & 15;
  const int q0   = blockIdx.x * 128;    // q-row offset within sequence
  const size_t rowbase = (size_t)bN * S_DIM;
  const int colh = h * D_DIM;

  // Stage Q tile: 128 rows x 64 cols (resident for whole kernel)
  {
    const int qr   = tid >> 1;
    const int cseg = (tid & 1) * 32;
    const __bf16* g = Q + (rowbase + q0 + qr) * E_DIM + colh + cseg;
#pragma unroll
    for (int j = 0; j < 4; ++j)
      *(v8bf*)(&Qs[qr * QSTR + cseg + j * 8]) = *(const v8bf*)(g + j * 8);
  }

  float mrow[8], lsum[8];
#pragma unroll
  for (int e = 0; e < 8; ++e) { mrow[e] = -1e30f; lsum[e] = 0.0f; }
  v8f oacc[4] = {};

  const int cl    = lane & 15;
  const int half8 = (lane >> 4) * 8;

  // staging indices for K/V tiles (64 rows x 64 cols)
  const int r   = tid >> 2;          // 0..63
  const int seg = (tid & 3) * 16;    // 0,16,32,48

  int buf = 0;
  v8bf vr0, vr1;
#if !USE_ASYNC
  v8bf kr0, kr1;
#endif

  // -------- prologue: stage key-block 0 into buffer 0 --------
  {
    const __bf16* gk = Kk + (rowbase + r) * E_DIM + colh + seg;
    const __bf16* gv = V  + (rowbase + r) * N_QKV + colh + seg;
#if USE_ASYNC
    async_copy16(gk,     &Ks[0][r * QSTR + seg]);
    async_copy16(gk + 8, &Ks[0][r * QSTR + seg + 8]);
#else
    kr0 = *(const v8bf*)(gk);
    kr1 = *(const v8bf*)(gk + 8);
    *(v8bf*)(&Ks[0][r * QSTR + seg])     = kr0;
    *(v8bf*)(&Ks[0][r * QSTR + seg + 8]) = kr1;
#endif
    vr0 = *(const v8bf*)(gv);
    vr1 = *(const v8bf*)(gv + 8);
#pragma unroll
    for (int j = 0; j < 8; ++j) {
      Vts[0][(seg + j    ) * QSTR + r] = vr0[j];
      Vts[0][(seg + 8 + j) * QSTR + r] = vr1[j];
    }
  }
  ASYNC_WAIT();
  __syncthreads();

  for (int kb = 0; kb < S_DIM; kb += 64) {
    const bool have_next = (kb + 64 < S_DIM);
    if (have_next) {
      const __bf16* gk = Kk + (rowbase + kb + 64 + r) * E_DIM + colh + seg;
      const __bf16* gv = V  + (rowbase + kb + 64 + r) * N_QKV + colh + seg;
#if USE_ASYNC
      async_copy16(gk,     &Ks[buf ^ 1][r * QSTR + seg]);
      async_copy16(gk + 8, &Ks[buf ^ 1][r * QSTR + seg + 8]);
#else
      kr0 = *(const v8bf*)(gk);
      kr1 = *(const v8bf*)(gk + 8);
#endif
      vr0 = *(const v8bf*)(gv);
      vr1 = *(const v8bf*)(gv + 8);
      if (kb + 128 < S_DIM) {
        __builtin_prefetch(gk + (size_t)64 * E_DIM, 0, 3);
        __builtin_prefetch(gv + (size_t)64 * N_QKV, 0, 3);
      }
    }

    // ---- scores: this wave's 16 q rows x 64 keys (4 C tiles) ----
    v8f s[4] = {};
#pragma unroll
    for (int ks = 0; ks < D_DIM; ks += 32) {
      v16bf aq = load_frag<QSTR>(&Qs[(wid * 16) * QSTR + ks], lane);
#pragma unroll
      for (int t = 0; t < 4; ++t) {
        v16bf bk = load_frag<QSTR>(&Ks[buf][(t * 16) * QSTR + ks], lane);
        s[t] = __builtin_amdgcn_wmma_f32_16x16x32_bf16(
            false, aq, false, bk, (short)0, s[t], false, false);
      }
    }

    // ---- online softmax; each row's 64 vals live in one half-wave ----
#pragma unroll
    for (int e = 0; e < 8; ++e) {
      float pmax = -1e30f;
#pragma unroll
      for (int t = 0; t < 4; ++t) pmax = fmaxf(pmax, s[t][e] * 0.125f);
      for (int msk = 1; msk < 16; msk <<= 1)
        pmax = fmaxf(pmax, __shfl_xor(pmax, msk));
      const float mnew  = fmaxf(mrow[e], pmax);
      const float alpha = __expf(mrow[e] - mnew);
      float psum = 0.0f;
#pragma unroll
      for (int t = 0; t < 4; ++t) {
        const float p = __expf(s[t][e] * 0.125f - mnew);
        s[t][e] = p;
        psum += p;
      }
      for (int msk = 1; msk < 16; msk <<= 1)
        psum += __shfl_xor(psum, msk);
      lsum[e] = lsum[e] * alpha + psum;
      mrow[e] = mnew;
#pragma unroll
      for (int t = 0; t < 4; ++t) oacc[t][e] *= alpha;
    }

    // ---- stage P (bf16) in wave-private LDS, re-read as A operand ----
#pragma unroll
    for (int t = 0; t < 4; ++t)
#pragma unroll
      for (int e = 0; e < 8; ++e)
        Ps[wid][(e + half8) * QSTR + t * 16 + cl] = f2bf(s[t][e]);

    // ---- O += P @ V  (K dim = 64 keys) ----
#pragma unroll
    for (int ks = 0; ks < 64; ks += 32) {
      v16bf ap = load_frag<QSTR>(&Ps[wid][ks], lane);
#pragma unroll
      for (int t = 0; t < 4; ++t) {
        v16bf bv = load_frag<QSTR>(&Vts[buf][(t * 16) * QSTR + ks], lane);
        oacc[t] = __builtin_amdgcn_wmma_f32_16x16x32_bf16(
            false, ap, false, bv, (short)0, oacc[t], false, false);
      }
    }

    // ---- refill the other buffer (read last iteration; safe now) ----
    if (have_next) {
#if !USE_ASYNC
      *(v8bf*)(&Ks[buf ^ 1][r * QSTR + seg])     = kr0;
      *(v8bf*)(&Ks[buf ^ 1][r * QSTR + seg + 8]) = kr1;
#endif
#pragma unroll
      for (int j = 0; j < 8; ++j) {
        Vts[buf ^ 1][(seg + j    ) * QSTR + r] = vr0[j];
        Vts[buf ^ 1][(seg + 8 + j) * QSTR + r] = vr1[j];
      }
    }
    ASYNC_WAIT();
    __syncthreads();
    buf ^= 1;
  }

  // ---- normalize and write out ----
#pragma unroll
  for (int e = 0; e < 8; ++e) lsum[e] = 1.0f / lsum[e];
#pragma unroll
  for (int t = 0; t < 4; ++t)
#pragma unroll
    for (int e = 0; e < 8; ++e) {
      const int row = q0 + wid * 16 + half8 + e;
      O[(rowbase + row) * E_DIM + colh + t * 16 + cl] =
          f2bf(oacc[t][e] * lsum[e]);
    }
}

// ---------------------------------------------------------------------------
// Host-side orchestration
// ---------------------------------------------------------------------------
extern "C" void kernel_launch(void* const* d_in, const int* in_sizes, int n_in,
                              void* d_out, int out_size, void* d_ws,
                              size_t ws_size, hipStream_t stream) {
  const float* x    = (const float*)d_in[0];
  const float* rot  = (const float*)d_in[1];
  const float* ent  = (const float*)d_in[2];
  const float* qkvw = (const float*)d_in[3];
  const float* qkvb = (const float*)d_in[4];
  const float* outw = (const float*)d_in[5];
  const float* outb = (const float*)d_in[6];
  float* out = (float*)d_out;

  char* ws = (char*)d_ws;
  const size_t MB = 1024u * 1024u;
  __bf16* xb   = (__bf16*)(ws);            //  8 MB  x in bf16
  __bf16* wqkv = (__bf16*)(ws +  8 * MB);  //  6 MB  qkv_w (already N x K)
  __bf16* rt   = (__bf16*)(ws + 14 * MB);  //  2 MB  rotation^T
  __bf16* et   = (__bf16*)(ws + 16 * MB);  //  2 MB  entangle^T
  __bf16* wout = (__bf16*)(ws + 18 * MB);  //  2 MB  out_w (already N x K)
  __bf16* qkv  = (__bf16*)(ws + 20 * MB);  // 24 MB  qkv activations
  __bf16* q2   = (__bf16*)(ws + 44 * MB);  //  8 MB  q @ rotation
  __bf16* k2   = (__bf16*)(ws + 52 * MB);  //  8 MB  k @ entangle
  __bf16* ao   = (__bf16*)(ws + 60 * MB);  //  8 MB  attention output

  int n;
  n = M_DIM * E_DIM;
  convert_bf16_kernel<<<(n / 8 + 255) / 256, 256, 0, stream>>>(x, xb, n);
  n = N_QKV * E_DIM;
  convert_bf16_kernel<<<(n / 8 + 255) / 256, 256, 0, stream>>>(qkvw, wqkv, n);
  n = E_DIM * E_DIM;
  convert_bf16_kernel<<<(n / 8 + 255) / 256, 256, 0, stream>>>(outw, wout, n);

  dim3 tg((E_DIM + 255) / 256, E_DIM);
  transpose_bf16_kernel<<<tg, 256, 0, stream>>>(rot, rt, E_DIM, E_DIM);
  transpose_bf16_kernel<<<tg, 256, 0, stream>>>(ent, et, E_DIM, E_DIM);

  const dim3 blk(256);
  // qkv = x @ qkv_w^T + qkv_b                     (4096 x 3072, K=1024)
  gemm_bf16<false><<<dim3(N_QKV / 128, M_DIM / 128), blk, 0, stream>>>(
      xb, E_DIM, wqkv, E_DIM, qkv, N_QKV, E_DIM, qkvb, nullptr);
  // q2 = q @ rotation                             (4096 x 1024, K=1024)
  gemm_bf16<false><<<dim3(E_DIM / 128, M_DIM / 128), blk, 0, stream>>>(
      qkv, N_QKV, rt, E_DIM, q2, E_DIM, E_DIM, nullptr, nullptr);
  // k2 = k @ entangle
  gemm_bf16<false><<<dim3(E_DIM / 128, M_DIM / 128), blk, 0, stream>>>(
      qkv + E_DIM, N_QKV, et, E_DIM, k2, E_DIM, E_DIM, nullptr, nullptr);
  // flash attention over heads
  attn_kernel<<<dim3(S_DIM / 128, B_DIM * H_DIM), blk, 0, stream>>>(
      q2, k2, qkv + 2 * E_DIM, ao);
  // out = ao @ out_w^T + out_b + x  (fp32 to d_out)
  gemm_bf16<true><<<dim3(E_DIM / 128, M_DIM / 128), blk, 0, stream>>>(
      ao, E_DIM, wout, E_DIM, out, E_DIM, E_DIM, outb, x);
}
